// GatedDeltaNetCanon_15487652069728
// MI455X (gfx1250) — matrix operand order repaired and checked by reading, not compiled, not run on hardware
//
#include <hip/hip_runtime.h>
#include <hip/hip_bf16.h>

typedef __bf16 bf16;
typedef __attribute__((ext_vector_type(16))) __bf16 v16bf;
typedef __attribute__((ext_vector_type(8)))  float  v8f;

#define NH    6
#define DKD   256
#define DVD   512
#define HIDD  2048
#define TLEN  4096
#define BSZ   2
#define MROWS (BSZ*TLEN)
#define CONVK 4

// ---------------------------------------------------------------------------
// CDNA5 async global->LDS copy (ASYNCcnt-tracked), guarded so the build can't
// break if the toolchain lacks the builtin; falls back to plain copies.
// Builtin signature (from hipcc diagnostic): arg0 = int4 addrspace(1)*,
// arg1 = LDS pointer, then two immediates (offset, cpol).
// ---------------------------------------------------------------------------
#if defined(__has_builtin)
#if __has_builtin(__builtin_amdgcn_global_load_async_to_lds_b128) && \
    __has_builtin(__builtin_amdgcn_s_wait_asynccnt)
#define USE_ASYNC_COPY 1
#endif
#endif
#ifndef USE_ASYNC_COPY
#define USE_ASYNC_COPY 0
#endif

#if USE_ASYNC_COPY
typedef int v4i_g __attribute__((vector_size(16)));
typedef __attribute__((address_space(1))) v4i_g as1_v4i;
typedef __attribute__((address_space(3))) v4i_g as3_v4i;
__device__ __forceinline__ void async_copy16(const void* g, void* l) {
  __builtin_amdgcn_global_load_async_to_lds_b128((as1_v4i*)g, (as3_v4i*)l, 0, 0);
}
#endif

// ---------------------------------------------------------------------------
// WMMA helpers (CDNA5 wave32, 16x16x32 bf16 -> f32)
// ---------------------------------------------------------------------------
__device__ __forceinline__ v8f wmma_bf16(v16bf a, v16bf b, v8f c) {
  return __builtin_amdgcn_wmma_f32_16x16x32_bf16(false, a, false, b, (short)0, c,
                                                 false, false);
}

// A-operand fragment from row-major [.., pitch] bf16 buffer.
// lanes 0-15 (row m0+lane): K = {k0..k0+7, k0+16..k0+23};
// lanes 16-31 (row m0+lane-16): K = {k0+8..k0+15, k0+24..k0+31}.
__device__ __forceinline__ v16bf fragA(const bf16* buf, int pitch, int m0, int k0,
                                       int lane) {
  int half = lane >> 4;
  const bf16* p = buf + (m0 + (lane & 15)) * pitch + k0 + half * 8;
  v16bf a;
#pragma unroll
  for (int i = 0; i < 8; i++) { a[i] = p[i]; a[8 + i] = p[16 + i]; }
  return a;
}

// B-operand fragment: source stores B transposed (row n = output column n,
// K contiguous). Lane n<16: K={k0..k0+7, k0+16..k0+23}, lane 16+n: K={k0+8..k0+15, k0+24..k0+31}.
__device__ __forceinline__ v16bf fragB(const bf16* buf, int pitch, int n0, int k0,
                                       int lane) {
  const bf16* p = buf + (n0 + (lane & 15)) * pitch + k0 + (lane >> 4) * 8;
  v16bf b;
#pragma unroll
  for (int i = 0; i < 8; i++) { b[i] = p[i]; b[8 + i] = p[16 + i]; }
  return b;
}

// ---------------------------------------------------------------------------
__global__ void cvt_bf16_kernel(const float* __restrict__ src, bf16* __restrict__ dst,
                                size_t n) {
  size_t i = (size_t)blockIdx.x * blockDim.x + threadIdx.x;
  size_t stride = (size_t)gridDim.x * blockDim.x;
  for (; i < n; i += stride) dst[i] = (bf16)src[i];
}

// ---------------------------------------------------------------------------
// GEMM: C[M,N] = A[M,K] * B[N,K]^T (A,B bf16 row-major, C fp32).
// Block tile 128x128, 256 threads (8 waves), wave tile 32x64 (2x4 WMMA).
// Double-buffered LDS; async global->LDS (ASYNCcnt) when available.
// ---------------------------------------------------------------------------
__launch_bounds__(256)
__global__ void gemm_bf16_nt(const bf16* __restrict__ A, const bf16* __restrict__ B,
                             float* __restrict__ C, int Mr, int Nc, int Kd) {
  __shared__ bf16 As[2][128][40];  // pitch 40 -> conflict-free fragment reads
  __shared__ bf16 Bs[2][128][40];
  int tid = threadIdx.x;
  int lane = tid & 31, w = tid >> 5;
  int bm = blockIdx.x * 128, bn = blockIdx.y * 128;
  int wm = (w & 3) * 32, wn = (w >> 2) * 64;
  v8f c[2][4] = {};

  // loader mapping: 128 rows x 32 K, 2 threads/row, 16 bf16 (32B) each
  int ra = tid >> 1, ca = (tid & 1) * 16;
  const bf16* ga = A + (size_t)(bm + ra) * Kd + ca;
  const bf16* gb = B + (size_t)(bn + ra) * Kd + ca;

  auto stage = [&](int buf, int k0) {
#if USE_ASYNC_COPY
    async_copy16(ga + k0, &As[buf][ra][ca]);
    async_copy16(ga + k0 + 8, &As[buf][ra][ca + 8]);
    async_copy16(gb + k0, &Bs[buf][ra][ca]);
    async_copy16(gb + k0 + 8, &Bs[buf][ra][ca + 8]);
#else
    uint4 a0 = *(const uint4*)(ga + k0);
    uint4 a1 = *(const uint4*)(ga + k0 + 8);
    uint4 b0 = *(const uint4*)(gb + k0);
    uint4 b1 = *(const uint4*)(gb + k0 + 8);
    *(uint4*)&As[buf][ra][ca] = a0;
    *(uint4*)&As[buf][ra][ca + 8] = a1;
    *(uint4*)&Bs[buf][ra][ca] = b0;
    *(uint4*)&Bs[buf][ra][ca + 8] = b1;
#endif
  };
  auto compute = [&](int cur) {
    v16bf a0 = fragA(&As[cur][0][0], 40, wm, 0, lane);
    v16bf a1 = fragA(&As[cur][0][0], 40, wm + 16, 0, lane);
#pragma unroll
    for (int j = 0; j < 4; j++) {
      v16bf bj = fragB(&Bs[cur][0][0], 40, wn + j * 16, 0, lane);
      c[0][j] = wmma_bf16(a0, bj, c[0][j]);
      c[1][j] = wmma_bf16(a1, bj, c[1][j]);
    }
  };

  int nk = Kd / 32;
  stage(0, 0);
  for (int kk = 0; kk < nk - 1; kk++) {
    int cur = kk & 1;
    stage(cur ^ 1, (kk + 1) * 32);
#if USE_ASYNC_COPY
    __builtin_amdgcn_s_wait_asynccnt(4);  // retire the 4 older (current-buf) loads
#endif
    __syncthreads();
    compute(cur);
    __syncthreads();
  }
#if USE_ASYNC_COPY
  __builtin_amdgcn_s_wait_asynccnt(0);
#endif
  __syncthreads();
  compute((nk - 1) & 1);

  int half = lane >> 4, nl = lane & 15;
#pragma unroll
  for (int i = 0; i < 2; i++)
#pragma unroll
    for (int j = 0; j < 4; j++)
#pragma unroll
      for (int r = 0; r < 8; r++)
        C[(size_t)(bm + wm + i * 16 + r + 8 * half) * Nc + bn + wn + j * 16 + nl] =
            c[i][j][r];
}

// ---------------------------------------------------------------------------
// beta / g head projections: one wave per (token, head)
// ---------------------------------------------------------------------------
__global__ void smallproj_kernel(const float* __restrict__ x, const float* __restrict__ Wb,
                                 const float* __restrict__ Wa, const float* __restrict__ A_log,
                                 const float* __restrict__ dt_bias,
                                 float* __restrict__ beta_arr, float* __restrict__ g_arr) {
  int wid = blockIdx.x * 8 + (threadIdx.x >> 5);
  int lane = threadIdx.x & 31;
  int m = wid / NH, h = wid % NH;
  const float* xr = x + (size_t)m * HIDD;
  const float* wb = Wb + (size_t)h * HIDD;
  const float* wa = Wa + (size_t)h * HIDD;
  float sb = 0.f, sa = 0.f;
  for (int i = lane; i < HIDD; i += 32) {
    float xv = xr[i];
    sb += xv * wb[i];
    sa += xv * wa[i];
  }
#pragma unroll
  for (int o = 16; o > 0; o >>= 1) {
    sb += __shfl_xor(sb, o, 32);
    sa += __shfl_xor(sa, o, 32);
  }
  if (lane == 0) {
    int b = m >> 12, t = m & (TLEN - 1);
    size_t idx = ((size_t)(b * NH + h)) * TLEN + t;
    beta_arr[idx] = 1.f / (1.f + __expf(-sb));
    float z = sa + dt_bias[h];
    float sp = (z > 20.f) ? z : log1pf(__expf(z));
    g_arr[idx] = -__expf(A_log[h]) * sp;
  }
}

// ---------------------------------------------------------------------------
// causal depthwise conv(4) + SiLU + L2 norm (q/k) -> bf16 [B,H,T,DK]
// ---------------------------------------------------------------------------
__global__ void conv_norm_qk_kernel(const float* __restrict__ pre,
                                    const float* __restrict__ convw,
                                    bf16* __restrict__ outh, float scale) {
  int m = blockIdx.x / NH, h = blockIdx.x % NH;
  int c = threadIdx.x;
  int t = m & (TLEN - 1), b = m >> 12;
  int cg = h * DKD + c;
  const float* wr = convw + (size_t)cg * CONVK;
  float v = 0.f;
#pragma unroll
  for (int j = 0; j < CONVK; j++) {
    int tt = t - 3 + j;
    if (tt >= 0) v += wr[j] * pre[(size_t)(m - 3 + j) * (NH * DKD) + cg];
  }
  v = v / (1.f + __expf(-v));  // SiLU
  __shared__ float red[256];
  red[c] = v * v;
  __syncthreads();
#pragma unroll
  for (int s = 128; s > 0; s >>= 1) {
    if (c < s) red[c] += red[c + s];
    __syncthreads();
  }
  float rs = rsqrtf(red[0] + 1e-6f) * scale;
  outh[(((size_t)(b * NH + h)) * TLEN + t) * DKD + c] = (bf16)(v * rs);
}

// causal conv + SiLU for v -> bf16 [B,H,T,DV]
__global__ void conv_v_kernel(const float* __restrict__ pre, const float* __restrict__ convw,
                              bf16* __restrict__ outh) {
  int m = blockIdx.x / NH, h = blockIdx.x % NH;
  int t = m & (TLEN - 1), b = m >> 12;
  for (int c = threadIdx.x; c < DVD; c += 256) {
    int cg = h * DVD + c;
    const float* wr = convw + (size_t)cg * CONVK;
    float v = 0.f;
#pragma unroll
    for (int j = 0; j < CONVK; j++) {
      int tt = t - 3 + j;
      if (tt >= 0) v += wr[j] * pre[(size_t)(m - 3 + j) * (NH * DVD) + cg];
    }
    v = v / (1.f + __expf(-v));
    outh[(((size_t)(b * NH + h)) * TLEN + t) * DVD + c] = (bf16)v;
  }
}

// ---------------------------------------------------------------------------
// Chunked gated delta-rule recurrence. Chunk C=64, DV slice = 64.
// grid = B*H*8 workgroups, 256 threads (8 waves). State S[256][64] fp32 in LDS.
// ---------------------------------------------------------------------------
__launch_bounds__(256)
__global__ void gdn_recurrence_kernel(const bf16* __restrict__ qh, const bf16* __restrict__ kh,
                                      const bf16* __restrict__ vh, const float* __restrict__ g_arr,
                                      const float* __restrict__ beta_arr,
                                      float* __restrict__ o_out) {
  __shared__ float Sf[256][68];     // persistent fp32 state slice
  __shared__ bf16 SbT[64][264];     // S transposed, bf16 (WMMA B operand)
  __shared__ bf16 Qs[64][264];
  __shared__ bf16 Ks[64][264];
  __shared__ bf16 KTs[256][72];     // K chunk transposed (dk-major)
  __shared__ bf16 Vs[64][72];
  __shared__ float Mlds[64][68];    // strict-lower attention matrix
  __shared__ bf16 AqBf[64][72];     // inclusive-lower q-k matrix, bf16
  __shared__ float PT[64][68];      // (beta*a*(K@S)) transposed
  __shared__ float UTf[64][68];     // U transposed, fp32 (solve)
  __shared__ bf16 UTbf[64][72];     // U transposed, bf16 (O GEMM)
  __shared__ bf16 UpTbf[64][72];    // (a_last/a_t)*U transposed (state update)
  __shared__ float gcs[64], asc[64], bts[64], upf[64];

  int tid = threadIdx.x, lane = tid & 31, w = tid >> 5;
  int wg = blockIdx.x;
  int slice = wg & 7, bh = wg >> 3;
  int dv0 = slice * 64;
  int bidx = bh / NH, hidx = bh % NH;
  const bf16* qp = qh + (size_t)bh * TLEN * DKD;
  const bf16* kp = kh + (size_t)bh * TLEN * DKD;
  const bf16* vp = vh + (size_t)bh * TLEN * DVD + dv0;
  const float* gp = g_arr + (size_t)bh * TLEN;
  const float* bp = beta_arr + (size_t)bh * TLEN;
  float* op = o_out + (size_t)bidx * TLEN * (NH * DVD) + hidx * DVD + dv0;

  for (int i = tid; i < 256 * 68; i += 256) (&Sf[0][0])[i] = 0.f;
  __syncthreads();

  for (int ch = 0; ch < TLEN / 64; ch++) {
    int t0 = ch * 64;
    {  // load Q,K chunk (vectorized 16B moves)
      int r = tid >> 2, cb = (tid & 3) * 64;
      const uint4* sq = (const uint4*)(qp + (size_t)(t0 + r) * DKD + cb);
      const uint4* sk = (const uint4*)(kp + (size_t)(t0 + r) * DKD + cb);
      uint4* dq = (uint4*)&Qs[r][cb];
      uint4* dk = (uint4*)&Ks[r][cb];
#pragma unroll
      for (int i = 0; i < 8; i++) { dq[i] = sq[i]; dk[i] = sk[i]; }
    }
    {  // load V slice (32B per thread)
      int r = tid >> 2, cb = (tid & 3) * 16;
      const uint4* sv = (const uint4*)(vp + (size_t)(t0 + r) * DVD + cb);
      uint4* dv4 = (uint4*)&Vs[r][cb];
      dv4[0] = sv[0];
      dv4[1] = sv[1];
    }
    if (tid < 64) { gcs[tid] = gp[t0 + tid]; bts[tid] = bp[t0 + tid]; }
    {  // bf16 transposed copy of state
      int dv = tid >> 2, kb = (tid & 3) * 64;
      for (int i = 0; i < 64; i++) SbT[dv][kb + i] = (bf16)Sf[kb + i][dv];
    }
    __syncthreads();
    {  // transpose K inside LDS (coalesced row reads, scattered b16 writes)
      int dk = tid;
      for (int t = 0; t < 64; t++) KTs[dk][t] = Ks[t][dk];
    }
    if (tid == 0) {  // inclusive cumsum of log-decay (all <= 0)
      float run = 0.f;
      for (int i = 0; i < 64; i++) { run += gcs[i]; gcs[i] = run; }
    }
    __syncthreads();
    if (tid < 64) { asc[tid] = __expf(gcs[tid]); upf[tid] = __expf(gcs[63] - gcs[tid]); }
    __syncthreads();

    // ---- G1: KK^T -> M,  QK^T -> Aq   (64x64 each, 2 tiles/wave) ----
    {
      v8f accK[2] = {}, accQ[2] = {};
#pragma unroll
      for (int kk = 0; kk < 8; kk++) {
        int k0 = kk * 32;
#pragma unroll
        for (int ti = 0; ti < 2; ti++) {
          int tile = w * 2 + ti;
          int mi = (tile >> 2) * 16, ni = (tile & 3) * 16;
          v16bf bK = fragB(&Ks[0][0], 264, ni, k0, lane);
          v16bf aK = fragA(&Ks[0][0], 264, mi, k0, lane);
          v16bf aQ = fragA(&Qs[0][0], 264, mi, k0, lane);
          accK[ti] = wmma_bf16(aK, bK, accK[ti]);
          accQ[ti] = wmma_bf16(aQ, bK, accQ[ti]);
        }
      }
      int half = lane >> 4, nl = lane & 15;
#pragma unroll
      for (int ti = 0; ti < 2; ti++) {
        int tile = w * 2 + ti;
        int mi = (tile >> 2) * 16, ni = (tile & 3) * 16;
        int j = ni + nl;
#pragma unroll
        for (int r = 0; r < 8; r++) {
          int t = mi + r + 8 * half;
          float e = __expf(fminf(gcs[t] - gcs[j], 0.f));
          Mlds[t][j] = (t > j) ? accK[ti][r] * bts[t] * e : 0.f;
          AqBf[t][j] = (bf16)((t >= j) ? accQ[ti][r] * e : 0.f);
        }
      }
    }

    // ---- G2: P = K@S (scaled beta*a), O1 = Q@S (scaled a) ----
    v8f oacc[2] = {};
    {
      v8f accP[2] = {};
#pragma unroll
      for (int kk = 0; kk < 8; kk++) {
        int k0 = kk * 32;
#pragma unroll
        for (int ti = 0; ti < 2; ti++) {
          int tile = w * 2 + ti;
          int mi = (tile >> 2) * 16, ni = (tile & 3) * 16;
          v16bf bS = fragB(&SbT[0][0], 264, ni, k0, lane);
          v16bf aK = fragA(&Ks[0][0], 264, mi, k0, lane);
          v16bf aQ = fragA(&Qs[0][0], 264, mi, k0, lane);
          accP[ti] = wmma_bf16(aK, bS, accP[ti]);
          oacc[ti] = wmma_bf16(aQ, bS, oacc[ti]);
        }
      }
      int half = lane >> 4, nl = lane & 15;
#pragma unroll
      for (int ti = 0; ti < 2; ti++) {
        int tile = w * 2 + ti;
        int mi = (tile >> 2) * 16, ni = (tile & 3) * 16;
        int dvc = ni + nl;
#pragma unroll
        for (int r = 0; r < 8; r++) {
          int t = mi + r + 8 * half;
          PT[dvc][t] = accP[ti][r] * bts[t] * asc[t];
          oacc[ti][r] *= asc[t];
        }
      }
    }
    __syncthreads();

    // ---- triangular solve: u_t = r_t - sum_{j<t} M[t][j] u_j ----
    if (tid < 64) {
      int dv = tid;
      for (int t = 0; t < 64; t++) {
        float acc = bts[t] * (float)Vs[t][dv] - PT[dv][t];
        for (int j = 0; j < t; j++) acc -= Mlds[t][j] * UTf[dv][j];
        UTf[dv][t] = acc;
        UTbf[dv][t] = (bf16)acc;
        UpTbf[dv][t] = (bf16)(acc * upf[t]);
      }
    }
    __syncthreads();

    // ---- G5: O += Aq @ U ; store chunk output ----
    {
#pragma unroll
      for (int kk = 0; kk < 2; kk++) {
        int k0 = kk * 32;
#pragma unroll
        for (int ti = 0; ti < 2; ti++) {
          int tile = w * 2 + ti;
          int mi = (tile >> 2) * 16, ni = (tile & 3) * 16;
          v16bf aA = fragA(&AqBf[0][0], 72, mi, k0, lane);
          v16bf bU = fragB(&UTbf[0][0], 72, ni, k0, lane);
          oacc[ti] = wmma_bf16(aA, bU, oacc[ti]);
        }
      }
      int half = lane >> 4, nl = lane & 15;
#pragma unroll
      for (int ti = 0; ti < 2; ti++) {
        int tile = w * 2 + ti;
        int mi = (tile >> 2) * 16, ni = (tile & 3) * 16;
#pragma unroll
        for (int r = 0; r < 8; r++) {
          int t = t0 + mi + r + 8 * half;
          op[(size_t)t * (NH * DVD) + ni + nl] = oacc[ti][r];
        }
      }
    }

    // ---- G6: S = a_last*S + K^T @ U'   (256x64, 8 tiles/wave) ----
    {
      float alast = asc[63];
      int half = lane >> 4, nl = lane & 15;
#pragma unroll
      for (int idx = 0; idx < 8; idx++) {
        int tile = w * 8 + idx;
        int m0 = (tile >> 2) * 16, n0 = (tile & 3) * 16;
        v8f c;
#pragma unroll
        for (int r = 0; r < 8; r++) c[r] = Sf[m0 + r + 8 * half][n0 + nl] * alast;
#pragma unroll
        for (int kk = 0; kk < 2; kk++) {
          v16bf aK = fragA(&KTs[0][0], 72, m0, kk * 32, lane);
          v16bf bU = fragB(&UpTbf[0][0], 72, n0, kk * 32, lane);
          c = wmma_bf16(aK, bU, c);
        }
#pragma unroll
        for (int r = 0; r < 8; r++) Sf[m0 + r + 8 * half][n0 + nl] = c[r];
      }
    }
    __syncthreads();
  }
}

// ---------------------------------------------------------------------------
// per-head RMSNorm * norm_w * SiLU(gate) -> bf16
// ---------------------------------------------------------------------------
__global__ void gate_norm_kernel(const float* __restrict__ o_in, const float* __restrict__ gate,
                                 const float* __restrict__ norm_w, bf16* __restrict__ ob) {
  int m = blockIdx.x / NH, h = blockIdx.x % NH;
  float vals[2];
  float ssum = 0.f;
#pragma unroll
  for (int i = 0; i < 2; i++) {
    int c = threadIdx.x + i * 256;
    vals[i] = o_in[(size_t)m * (NH * DVD) + h * DVD + c];
    ssum += vals[i] * vals[i];
  }
  __shared__ float red[256];
  red[threadIdx.x] = ssum;
  __syncthreads();
#pragma unroll
  for (int s = 128; s > 0; s >>= 1) {
    if (threadIdx.x < s) red[threadIdx.x] += red[threadIdx.x + s];
    __syncthreads();
  }
  float rs = rsqrtf(red[0] / (float)DVD + 1e-5f);
#pragma unroll
  for (int i = 0; i < 2; i++) {
    int c = threadIdx.x + i * 256;
    float gt = gate[(size_t)m * (NH * DVD) + h * DVD + c];
    float val = vals[i] * rs * norm_w[c] * (gt / (1.f + __expf(-gt)));
    ob[(size_t)m * (NH * DVD) + h * DVD + c] = (bf16)val;
  }
}

// ---------------------------------------------------------------------------
extern "C" void kernel_launch(void* const* d_in, const int* in_sizes, int n_in,
                              void* d_out, int out_size, void* d_ws, size_t ws_size,
                              hipStream_t stream) {
  const float* x       = (const float*)d_in[0];
  const float* Wq      = (const float*)d_in[1];
  const float* Wk      = (const float*)d_in[2];
  const float* Wv      = (const float*)d_in[3];
  const float* Wb      = (const float*)d_in[4];
  const float* Wa      = (const float*)d_in[5];
  const float* A_log   = (const float*)d_in[6];
  const float* dt_bias = (const float*)d_in[7];
  const float* conv_wq = (const float*)d_in[8];
  const float* conv_wk = (const float*)d_in[9];
  const float* conv_wv = (const float*)d_in[10];
  const float* Wg      = (const float*)d_in[11];
  const float* norm_w  = (const float*)d_in[12];
  const float* Wo      = (const float*)d_in[13];
  float* out = (float*)d_out;

  const size_t M = MROWS;
  char* ws = (char*)d_ws;
  size_t off = 0;
  auto alloc = [&](size_t bytes) -> char* {
    char* p = ws + off;
    off += (bytes + 255) & ~(size_t)255;
    return p;
  };
  bf16* xb   = (bf16*)alloc(M * HIDD * 2);
  bf16* Wqb  = (bf16*)alloc((size_t)NH * DKD * HIDD * 2);
  bf16* Wkb  = (bf16*)alloc((size_t)NH * DKD * HIDD * 2);
  bf16* Wvb  = (bf16*)alloc((size_t)NH * DVD * HIDD * 2);
  bf16* Wgb  = (bf16*)alloc((size_t)NH * DVD * HIDD * 2);
  bf16* Wob  = (bf16*)alloc((size_t)HIDD * NH * DVD * 2);
  float* qpre = (float*)alloc(M * (NH * DKD) * 4);
  float* kpre = (float*)alloc(M * (NH * DKD) * 4);
  float* vpre = (float*)alloc(M * (NH * DVD) * 4);
  float* gatb = (float*)alloc(M * (NH * DVD) * 4);
  float* beta = (float*)alloc((size_t)BSZ * NH * TLEN * 4);
  float* gdec = (float*)alloc((size_t)BSZ * NH * TLEN * 4);
  bf16* qhb  = (bf16*)alloc((size_t)BSZ * NH * TLEN * DKD * 2);
  bf16* khb  = (bf16*)alloc((size_t)BSZ * NH * TLEN * DKD * 2);
  bf16* vhb  = (bf16*)alloc((size_t)BSZ * NH * TLEN * DVD * 2);
  float* obuf = (float*)alloc(M * (NH * DVD) * 4);
  bf16* obf  = (bf16*)alloc(M * (NH * DVD) * 2);

  // 1) convert activations + weights to bf16
  cvt_bf16_kernel<<<4096, 256, 0, stream>>>(x, xb, M * HIDD);
  cvt_bf16_kernel<<<2048, 256, 0, stream>>>(Wq, Wqb, (size_t)NH * DKD * HIDD);
  cvt_bf16_kernel<<<2048, 256, 0, stream>>>(Wk, Wkb, (size_t)NH * DKD * HIDD);
  cvt_bf16_kernel<<<2048, 256, 0, stream>>>(Wv, Wvb, (size_t)NH * DVD * HIDD);
  cvt_bf16_kernel<<<2048, 256, 0, stream>>>(Wg, Wgb, (size_t)NH * DVD * HIDD);
  cvt_bf16_kernel<<<2048, 256, 0, stream>>>(Wo, Wob, (size_t)HIDD * NH * DVD);

  // 2) big projections (WMMA GEMMs, 128x128 tiles)
  gemm_bf16_nt<<<dim3(M / 128, (NH * DKD) / 128), 256, 0, stream>>>(xb, Wqb, qpre, M, NH * DKD, HIDD);
  gemm_bf16_nt<<<dim3(M / 128, (NH * DKD) / 128), 256, 0, stream>>>(xb, Wkb, kpre, M, NH * DKD, HIDD);
  gemm_bf16_nt<<<dim3(M / 128, (NH * DVD) / 128), 256, 0, stream>>>(xb, Wvb, vpre, M, NH * DVD, HIDD);
  gemm_bf16_nt<<<dim3(M / 128, (NH * DVD) / 128), 256, 0, stream>>>(xb, Wgb, gatb, M, NH * DVD, HIDD);

  // 3) beta / g projections
  smallproj_kernel<<<(M * NH) / 8, 256, 0, stream>>>(x, Wb, Wa, A_log, dt_bias, beta, gdec);

  // 4) conv + SiLU (+ L2 norm for q/k)
  conv_norm_qk_kernel<<<M * NH, 256, 0, stream>>>(qpre, conv_wq, qhb, 0.0625f);  // DK^-0.5
  conv_norm_qk_kernel<<<M * NH, 256, 0, stream>>>(kpre, conv_wk, khb, 1.0f);
  conv_v_kernel<<<M * NH, 256, 0, stream>>>(vpre, conv_wv, vhb);

  // 5) chunked gated delta-rule recurrence (WMMA)
  gdn_recurrence_kernel<<<BSZ * NH * 8, 256, 0, stream>>>(qhb, khb, vhb, gdec, beta, obuf);

  // 6) RMSNorm + SiLU gate
  gate_norm_kernel<<<M * NH, 256, 0, stream>>>(obuf, gatb, norm_w, obf);

  // 7) output projection (WMMA GEMM) -> d_out
  gemm_bf16_nt<<<dim3(M / 128, HIDD / 128), 256, 0, stream>>>(obf, Wob, out, M, HIDD, NH * DVD);
}
